// SoundCompassFusionFeature_39049842655270
// MI455X (gfx1250) — matrix operand order repaired and checked by reading, not compiled, not run on hardware
//
#include <hip/hip_runtime.h>
#include <cmath>

#define NBANDS 31
#define HCH    96
#define CIN    64
#define FBINS  257
#define TLEN   1000
#define BSZ    4

typedef __attribute__((ext_vector_type(16))) _Float16 v16h;
typedef __attribute__((ext_vector_type(8)))  _Float16 v8h;
typedef __attribute__((ext_vector_type(8)))  float    v8f;

struct BandTab { int s[NBANDS]; int e[NBANDS]; };

// ---------------------------------------------------------------------------
// WMMA helper (gfx1250 wave32, v_wmma_f32_16x16x32_f16)
// ---------------------------------------------------------------------------
__device__ __forceinline__ v8f wmma_f16(v16h a, v16h b, v8f c) {
  return __builtin_amdgcn_wmma_f32_16x16x32_f16(false, a, false, b, (short)0, c,
                                                false, false);
}

// A-matrix 16x32 f16 layout: lane m = l&15, half h = l>>4
//   element j in 0..15 -> K = kb + j + 16*h
// Weights pre-converted to f16 row-major (m, k); both 16B chunks aligned since
// ldk in {64,96} and kb,16*h are multiples of 8 halfs.
__device__ __forceinline__ v16h load_a_frag_h(const _Float16* W, int ldk,
                                              int mt, int kb, int lane) {
  int m = mt * 16 + (lane & 15);
  int h = lane >> 4;
  const _Float16* row = W + (size_t)m * ldk + kb + 16 * h;
  v8h lo = *(const v8h*)row;
  v8h hi = *(const v8h*)(row + 8);
  return __builtin_shufflevector(lo, hi, 0, 1, 2, 3, 4, 5, 6, 7, 8, 9, 10, 11,
                                 12, 13, 14, 15);
}

// B-matrix 32x16 f16 layout: col n = l&15, half h = l>>4, element j -> K = kb+j+16*h
// Source: global f32, element (k,n) at base[k*strideK + tcol]
__device__ __forceinline__ v16h load_b_frag_g(const float* base, size_t strideK,
                                              int kb, int tcol, int h) {
  v16h bf;
#pragma unroll
  for (int j = 0; j < 16; ++j)
    bf[j] = (_Float16)base[(size_t)(kb + 16 * h + j) * strideK + tcol];
  return bf;
}

// B fragment from per-wave LDS staging tile laid out [n][channel] (96 f16 per col)
__device__ __forceinline__ v16h load_b_frag_lds(const _Float16* stage, int kf,
                                                int lane) {
  int n = lane & 15, h = lane >> 4;
  const _Float16* p = stage + n * HCH + kf * 32 + 16 * h;
  v8h lo = *(const v8h*)p;
  v8h hi = *(const v8h*)(p + 8);
  return __builtin_shufflevector(lo, hi, 0, 1, 2, 3, 4, 5, 6, 7, 8, 9, 10, 11,
                                 12, 13, 14, 15);
}

// Channel LayerNorm over 96 channels split across half-wave pair (48 per lane)
__device__ __forceinline__ void chan_ln48(float* v) {
  float s = 0.f, s2 = 0.f;
#pragma unroll
  for (int i = 0; i < 48; ++i) { s += v[i]; s2 += v[i] * v[i]; }
  s  += __shfl_xor(s, 16);
  s2 += __shfl_xor(s2, 16);
  float mean = s * (1.0f / 96.0f);
  float var  = s2 * (1.0f / 96.0f) - mean * mean;
  float r = rsqrtf(var + 1e-5f);
#pragma unroll
  for (int i = 0; i < 48; ++i) v[i] = (v[i] - mean) * r;
}

// ---------------------------------------------------------------------------
// Kernel 0: f32 -> f16 weight pre-conversion (runs once per launch; weights
// are reused ~65k times by the fused kernel, so convert once, keep in L2)
// ---------------------------------------------------------------------------
__global__ void cvt_f16_kernel(const float* __restrict__ src,
                               _Float16* __restrict__ dst, int n) {
  int i = blockIdx.x * 256 + threadIdx.x;
  if (i < n) dst[i] = (_Float16)src[i];
}

// ---------------------------------------------------------------------------
// Kernel 1: FiLM gates  qp = doa @ dqg_w[k]^T + dqg_b[k]
// ---------------------------------------------------------------------------
__global__ void gates_kernel(const float* __restrict__ doa,
                             const float* __restrict__ dqg_w,
                             const float* __restrict__ dqg_b,
                             float* __restrict__ g1, float* __restrict__ g0) {
  int k = blockIdx.x, b = blockIdx.y, j = threadIdx.x;  // j in [0,192)
  const float* w = dqg_w + ((size_t)k * 2 * HCH + j) * 36;
  float acc = dqg_b[(size_t)k * 2 * HCH + j];
  for (int d = 0; d < 36; ++d) acc += doa[b * 36 + d] * w[d];
  if (j < HCH) g1[((size_t)k * BSZ + b) * HCH + j] = 1.0f + acc;
  else         g0[((size_t)k * BSZ + b) * HCH + (j - HCH)] = acc;
}

// ---------------------------------------------------------------------------
// Kernel 2: fully fused band-merge + attention. One wave = one (b, f, 16 t's).
// ---------------------------------------------------------------------------
__global__ __launch_bounds__(256) void fused_kernel(
    const float* __restrict__ mix, const float* __restrict__ spin,
    const _Float16* __restrict__ enc_wh, const float* __restrict__ enc_b,
    const float* __restrict__ enc_a, const _Float16* __restrict__ dec_wh,
    const float* __restrict__ dec_b, const float* __restrict__ dec_a,
    const _Float16* __restrict__ qwh, const float* __restrict__ qb,
    const _Float16* __restrict__ kwh, const float* __restrict__ kbias,
    const _Float16* __restrict__ vwh, const float* __restrict__ vb,
    const _Float16* __restrict__ owh, const float* __restrict__ ob,
    const float* __restrict__ g1, const float* __restrict__ g0,
    float* __restrict__ out, BandTab bt) {
  const int bf = blockIdx.x;
  const int b = bf / FBINS, f = bf % FBINS;
  const int wave = threadIdx.x >> 5, lane = threadIdx.x & 31;
  const int tile = blockIdx.y * 8 + wave;
  const int t0 = tile * 16;
  if (t0 >= TLEN) return;  // wave-uniform exit

  const int h = lane >> 4, nloc = lane & 15;
  int tcol = t0 + nloc;
  const bool tvalid = tcol < TLEN;
  if (!tvalid) tcol = TLEN - 1;  // clamp loads; stores masked

  __shared__ alignas(16) _Float16 ldsT[8][16 * HCH];  // per-wave staging
  _Float16* stage = &ldsT[wave][0];

  const size_t strideC = (size_t)FBINS * TLEN;
  const float* mix_b = mix + (size_t)b * HCH * strideC + (size_t)f * TLEN;

  // Prefetch the attention-phase mix_repr rows now (global_prefetch_b8):
  // 3 prefetches/lane cover all 96 strided rows; HBM fill overlaps band loop.
  {
    const float* pf = mix_b + (size_t)lane * strideC + (size_t)t0;
    __builtin_prefetch(pf, 0, 3);
    __builtin_prefetch(pf + 32 * strideC, 0, 3);
    __builtin_prefetch(pf + 64 * strideC, 0, 3);
  }

  // ---- spin B fragments (K = 64), loaded once, reused over covering bands
  const float* spin_b = spin + (size_t)b * CIN * strideC + (size_t)f * TLEN;
  v16h sb0 = load_b_frag_g(spin_b, strideC, 0, tcol, h);
  v16h sb1 = load_b_frag_g(spin_b, strideC, 32, tcol, h);

  float macc[48];  // merged accumulator, element (mt,i) -> macc[mt*8+i]
#pragma unroll
  for (int i = 0; i < 48; ++i) macc[i] = 0.f;
  float wsumf = 0.f;

  // ---- band loop (uniform per block: all waves share f)
  for (int k = 0; k < NBANDS; ++k) {
    const int s = bt.s[k], e = bt.e[k];
    if (f < s || f >= e) continue;
    const int L = e - s;
    const float win =
        0.5f - 0.5f * __cosf(6.28318530717958647692f * (float)(f - s) / (float)L);
    wsumf += win;

    // encoder GEMM: feat(96,16) = enc_w[k](96,64) x sub(64,16)
    const _Float16* Wenc = enc_wh + (size_t)k * HCH * CIN;
    float feat[48];
#pragma unroll
    for (int mt = 0; mt < 6; ++mt) {
      v8f c = {};
      c = wmma_f16(load_a_frag_h(Wenc, CIN, mt, 0, lane), sb0, c);
      c = wmma_f16(load_a_frag_h(Wenc, CIN, mt, 32, lane), sb1, c);
#pragma unroll
      for (int i = 0; i < 8; ++i)
        feat[mt * 8 + i] = c[i] + enc_b[(size_t)k * HCH + mt * 16 + i + 8 * h];
    }
    chan_ln48(feat);

    // PReLU + FiLM gate, stage to LDS as f16 B-operand
    const float ea = enc_a[k];
    const float* G1 = g1 + ((size_t)k * BSZ + b) * HCH;
    const float* G0 = g0 + ((size_t)k * BSZ + b) * HCH;
#pragma unroll
    for (int mt = 0; mt < 6; ++mt)
#pragma unroll
      for (int i = 0; i < 8; ++i) {
        int m = mt * 16 + i + 8 * h;
        float x = feat[mt * 8 + i];
        x = x >= 0.f ? x : ea * x;
        x = G1[m] * x + G0[m];
        stage[nloc * HCH + m] = (_Float16)x;
      }
    v16h tb0 = load_b_frag_lds(stage, 0, lane);
    v16h tb1 = load_b_frag_lds(stage, 1, lane);
    v16h tb2 = load_b_frag_lds(stage, 2, lane);

    // decoder GEMM: dec(96,16) = dec_w[k](96,96) x tgt(96,16)
    const _Float16* Wdec = dec_wh + (size_t)k * HCH * HCH;
    float dst[48];
#pragma unroll
    for (int mt = 0; mt < 6; ++mt) {
      v8f c = {};
      c = wmma_f16(load_a_frag_h(Wdec, HCH, mt, 0, lane), tb0, c);
      c = wmma_f16(load_a_frag_h(Wdec, HCH, mt, 32, lane), tb1, c);
      c = wmma_f16(load_a_frag_h(Wdec, HCH, mt, 64, lane), tb2, c);
#pragma unroll
      for (int i = 0; i < 8; ++i)
        dst[mt * 8 + i] = c[i] + dec_b[(size_t)k * HCH + mt * 16 + i + 8 * h];
    }
    chan_ln48(dst);
    const float da = dec_a[k];
#pragma unroll
    for (int i = 0; i < 48; ++i) {
      float x = dst[i];
      x = x >= 0.f ? x : da * x;
      macc[i] += win * x;
    }
  }

  // overlap-add normalization (merged stays in registers — never hits HBM)
  const float inv_w = 1.0f / fmaxf(wsumf, 1e-8f);
#pragma unroll
  for (int i = 0; i < 48; ++i) macc[i] *= inv_w;

  // stage merged for K/V B-operands
#pragma unroll
  for (int mt = 0; mt < 6; ++mt)
#pragma unroll
    for (int i = 0; i < 8; ++i)
      stage[nloc * HCH + mt * 16 + i + 8 * h] = (_Float16)macc[mt * 8 + i];
  v16h mb0 = load_b_frag_lds(stage, 0, lane);
  v16h mb1 = load_b_frag_lds(stage, 1, lane);
  v16h mb2 = load_b_frag_lds(stage, 2, lane);

  // mix_repr B fragments (K = 96) — lines were prefetched at kernel entry
  v16h xb0 = load_b_frag_g(mix_b, strideC, 0, tcol, h);
  v16h xb1 = load_b_frag_g(mix_b, strideC, 32, tcol, h);
  v16h xb2 = load_b_frag_g(mix_b, strideC, 64, tcol, h);

  // Q = qw@mix + qb, K = kw@merged + kb; score = sum_c Q*K (per column)
  float score = 0.f;
#pragma unroll
  for (int mt = 0; mt < 6; ++mt) {
    v8f cq = {};
    cq = wmma_f16(load_a_frag_h(qwh, HCH, mt, 0, lane), xb0, cq);
    cq = wmma_f16(load_a_frag_h(qwh, HCH, mt, 32, lane), xb1, cq);
    cq = wmma_f16(load_a_frag_h(qwh, HCH, mt, 64, lane), xb2, cq);
    v8f ck = {};
    ck = wmma_f16(load_a_frag_h(kwh, HCH, mt, 0, lane), mb0, ck);
    ck = wmma_f16(load_a_frag_h(kwh, HCH, mt, 32, lane), mb1, ck);
    ck = wmma_f16(load_a_frag_h(kwh, HCH, mt, 64, lane), mb2, ck);
#pragma unroll
    for (int i = 0; i < 8; ++i) {
      int m = mt * 16 + i + 8 * h;
      score += (cq[i] + qb[m]) * (ck[i] + kbias[m]);
    }
  }
  score += __shfl_xor(score, 16);
  const float attn =
      1.0f / (1.0f + __expf(-score * 0.10206207261596577f));  // 1/sqrt(96)

  // V = vw@merged + vb, attended = V*attn, restage as f16 B-operand
#pragma unroll
  for (int mt = 0; mt < 6; ++mt) {
    v8f cv = {};
    cv = wmma_f16(load_a_frag_h(vwh, HCH, mt, 0, lane), mb0, cv);
    cv = wmma_f16(load_a_frag_h(vwh, HCH, mt, 32, lane), mb1, cv);
    cv = wmma_f16(load_a_frag_h(vwh, HCH, mt, 64, lane), mb2, cv);
#pragma unroll
    for (int i = 0; i < 8; ++i) {
      int m = mt * 16 + i + 8 * h;
      stage[nloc * HCH + m] = (_Float16)((cv[i] + vb[m]) * attn);
    }
  }
  v16h ab0 = load_b_frag_lds(stage, 0, lane);
  v16h ab1 = load_b_frag_lds(stage, 1, lane);
  v16h ab2 = load_b_frag_lds(stage, 2, lane);

  // out = mix + ow@attended + ob
  float* outp = out + (size_t)b * HCH * strideC + (size_t)f * TLEN;
#pragma unroll
  for (int mt = 0; mt < 6; ++mt) {
    v8f co = {};
    co = wmma_f16(load_a_frag_h(owh, HCH, mt, 0, lane), ab0, co);
    co = wmma_f16(load_a_frag_h(owh, HCH, mt, 32, lane), ab1, co);
    co = wmma_f16(load_a_frag_h(owh, HCH, mt, 64, lane), ab2, co);
    if (tvalid) {
#pragma unroll
      for (int i = 0; i < 8; ++i) {
        int m = mt * 16 + i + 8 * h;
        size_t off = (size_t)m * strideC + (size_t)(t0 + nloc);
        outp[off] = mix_b[off] + co[i] + ob[m];
      }
    }
  }
}

// ---------------------------------------------------------------------------
extern "C" void kernel_launch(void* const* d_in, const int* in_sizes, int n_in,
                              void* d_out, int out_size, void* d_ws,
                              size_t ws_size, hipStream_t stream) {
  (void)in_sizes; (void)n_in; (void)out_size; (void)ws_size;
  const float* mix   = (const float*)d_in[0];
  const float* spin  = (const float*)d_in[1];
  const float* doa   = (const float*)d_in[2];
  const float* enc_w = (const float*)d_in[3];
  const float* enc_b = (const float*)d_in[4];
  const float* enc_a = (const float*)d_in[5];
  const float* dqg_w = (const float*)d_in[6];
  const float* dqg_b = (const float*)d_in[7];
  const float* dec_w = (const float*)d_in[8];
  const float* dec_b = (const float*)d_in[9];
  const float* dec_a = (const float*)d_in[10];
  const float* qw = (const float*)d_in[11];
  const float* qb = (const float*)d_in[12];
  const float* kw = (const float*)d_in[13];
  const float* kb = (const float*)d_in[14];
  const float* vw = (const float*)d_in[15];
  const float* vb = (const float*)d_in[16];
  const float* ow = (const float*)d_in[17];
  const float* ob = (const float*)d_in[18];
  float* out = (float*)d_out;

  // ---- workspace layout ----
  float* g1 = (float*)d_ws;                       // [31][4][96] f32
  float* g0 = g1 + (size_t)NBANDS * BSZ * HCH;    // [31][4][96] f32
  _Float16* enc_wh = (_Float16*)(g0 + (size_t)NBANDS * BSZ * HCH);
  const int n_enc = NBANDS * HCH * CIN;           // 190464
  _Float16* dec_wh = enc_wh + n_enc;
  const int n_dec = NBANDS * HCH * HCH;           // 285696
  _Float16* qwh = dec_wh + n_dec;
  _Float16* kwh = qwh + HCH * HCH;
  _Float16* vwh = kwh + HCH * HCH;
  _Float16* owh = vwh + HCH * HCH;

  // Band partition (mirrors the reference's numpy math; deterministic host calc)
  BandTab bt;
  {
    const double f_min = 16000.0 / 512.0, f_max = 8000.0;
    const double z_min = 69.0 + 12.0 * std::log2(f_min / 440.0);
    const double z_max = 69.0 + 12.0 * std::log2(f_max / 440.0);
    long long bins[NBANDS + 2];
    for (int i = 0; i < NBANDS + 2; ++i) {
      double z = z_min + (z_max - z_min) * (double)i / (double)(NBANDS + 1);
      double fp = 440.0 * std::pow(2.0, (z - 69.0) / 12.0);
      bins[i] = (long long)std::nearbyint(fp * 512.0 / 16000.0);
    }
    bins[0] = 0;
    for (int k = 0; k < NBANDS; ++k) {
      int s = (int)bins[k]; if (s < 0) s = 0;
      int e = (int)bins[k + 2] + 1; if (e > FBINS) e = FBINS;
      if (e <= s) e = s + 1;
      bt.s[k] = s; bt.e[k] = e;
    }
  }

  // weight pre-conversion (tiny; L2-resident afterwards)
  cvt_f16_kernel<<<(n_enc + 255) / 256, 256, 0, stream>>>(enc_w, enc_wh, n_enc);
  cvt_f16_kernel<<<(n_dec + 255) / 256, 256, 0, stream>>>(dec_w, dec_wh, n_dec);
  cvt_f16_kernel<<<(HCH * HCH + 255) / 256, 256, 0, stream>>>(qw, qwh, HCH * HCH);
  cvt_f16_kernel<<<(HCH * HCH + 255) / 256, 256, 0, stream>>>(kw, kwh, HCH * HCH);
  cvt_f16_kernel<<<(HCH * HCH + 255) / 256, 256, 0, stream>>>(vw, vwh, HCH * HCH);
  cvt_f16_kernel<<<(HCH * HCH + 255) / 256, 256, 0, stream>>>(ow, owh, HCH * HCH);
  gates_kernel<<<dim3(NBANDS, BSZ), 192, 0, stream>>>(doa, dqg_w, dqg_b, g1, g0);

  fused_kernel<<<dim3(BSZ * FBINS, 8), 256, 0, stream>>>(
      mix, spin, enc_wh, enc_b, enc_a, dec_wh, dec_b, dec_a, qwh, qb, kwh, kb,
      vwh, vb, owh, ob, g1, g0, out, bt);
}
